// S5SSM_16655883174608
// MI455X (gfx1250) — matrix operand order repaired and checked by reading, not compiled, not run on hardware
//
#include <hip/hip_runtime.h>
#include <hip/hip_bf16.h>
#include <math.h>

// ---------------------------------------------------------------------------
// S5 SSM forward for MI455X (gfx1250, wave32, WMMA + TDM async tile loads).
//   GEMM1:  Bu[m][n] = sum_k u_bf[m][k] * W1t[n][k]      (bf16 WMMA, f32 acc)
//   scan :  x_t = Lbar * x_{t-1} + Bu_t   (constant complex coeff, 3-pass)
//   GEMM2:  y[m][h]  = sum_k xs_bf[m][k] * W2t[h][k] + D[h]*u ; exact GELU
// GEMM tiles are DMA'd global->LDS by the Tensor Data Mover (tensor_load_to_lds,
// double-buffered, TENSORcnt-tracked) with hardware row padding to keep all
// ds_load_b128 fragment reads 16B-aligned and bank-spread.
// ---------------------------------------------------------------------------

#define B_SZ  8
#define HDIM  512
#define PDIM  256
#define LDIM  8192
#define K2    512          // = 2*PDIM = HDIM (reduction depth of both GEMMs)
#define LC    128          // scan segment length
#define NSEG  64           // LDIM / LC

typedef __attribute__((ext_vector_type(16))) __bf16         v16bf;
typedef __attribute__((ext_vector_type(8)))  float          v8f;
typedef __attribute__((ext_vector_type(8)))  unsigned short u16x8;
typedef __attribute__((ext_vector_type(4)))  unsigned int   u32x4;
typedef __attribute__((ext_vector_type(8)))  unsigned int   u32x8;

union FragBF { v16bf v; u16x8 h[2]; };

__device__ __forceinline__ unsigned short f2bf(float x) {
  unsigned int u = __builtin_bit_cast(unsigned int, x);
  u += 0x7fffu + ((u >> 16) & 1u);               // round-to-nearest-even
  return (unsigned short)(u >> 16);
}

// ---------------------------------------------------------------------------
// Workspace layout (bytes, all 256-aligned).  Total ~259 MB.
// ---------------------------------------------------------------------------
static const size_t OFF_UBF   = 0;                               // (B*L, H) bf16   64 MB
static const size_t OFF_W1T   = OFF_UBF  + (size_t)B_SZ*LDIM*HDIM*2;   // (2P, H) bf16
static const size_t OFF_W2T   = OFF_W1T  + (size_t)K2*HDIM*2;          // (H, 2P) bf16
static const size_t OFF_LAM   = OFF_W2T  + (size_t)HDIM*K2*2;          // (P,2) f32
static const size_t OFF_APOW  = OFF_LAM  + (size_t)PDIM*2*4;           // (P,2) f32
static const size_t OFF_F     = OFF_APOW + (size_t)PDIM*2*4;           // (P,2) f32
static const size_t OFF_CARRY = OFF_F    + (size_t)PDIM*2*4;           // (B,P,NSEG,2) f32
static const size_t OFF_BU    = OFF_CARRY+ (size_t)B_SZ*PDIM*NSEG*2*4; // (B*L, 2P) f32  128 MB
static const size_t OFF_XSBF  = OFF_BU   + (size_t)B_SZ*LDIM*K2*4;     // (B*L, 2P) bf16  64 MB

// ---------------------------------------------------------------------------
// Tensor Data Mover helpers (CDNA5 tensor_load_to_lds, VIMAGE encoding).
//
// D# group1 (constant per tile shape): data_size=2B, pad_enable=1,
//   pad_interval=3 (pad every 16 DWORDs = one 64 B row),
//   pad_amount=3 (4 DWORDs = 16 B)  ->  LDS row stride 80 B = 40 elements.
//   tensor_dim0 = 512, tile_dim0 = 32, tensor_dim0_stride = 512 elements.
// D# group0 (per issue): count=1, lds byte address, 57-bit global addr, type=2.
// ---------------------------------------------------------------------------
__device__ __forceinline__ u32x8 tdm_group1(unsigned tile_rows) {
  u32x8 g;
  g[0] = (1u << 16)      // data_size = 1 -> 2 bytes
       | (1u << 20)      // pad_enable
       | (3u << 22)      // pad_interval: 16 DWORDs (one 64B row)
       | (3u << 25);     // pad_amount: 4 DWORDs (16 B)
  g[1] = ((unsigned)K2 & 0xFFFFu) << 16;   // tensor_dim0 low16 (=512)
  g[2] = 0u;                               // tensor_dim0 hi | tensor_dim1 low16
  g[3] = 0x0010u | (32u << 16);            // tensor_dim1 hi (1M rows), tile_dim0=32
  g[4] = tile_rows;                        // tile_dim1 (tile_dim2 = 0 -> 2D)
  g[5] = (unsigned)K2;                     // tensor_dim0_stride low32 (=512)
  g[6] = 0u;
  g[7] = 0u;
  return g;
}

__device__ __forceinline__ void tdm_issue(const void* gptr, unsigned lds_byte,
                                          u32x8 g1) {
  unsigned long long ga = (unsigned long long)(uintptr_t)gptr;
  u32x4 g0;
  g0.x = 1u;                                    // count=1, user descriptor
  g0.y = lds_byte;                              // LDS destination (bytes)
  g0.z = (unsigned)ga;                          // global addr [31:0]
  g0.w = ((unsigned)(ga >> 32) & 0x01FFFFFFu)   // global addr [56:32]
       | (2u << 30);                            // type = 2 ("image")
  asm volatile("tensor_load_to_lds %0, %1" :: "s"(g0), "s"(g1) : "memory");
}

// ---------------------------------------------------------------------------
// k_prep_small: per-p scalars: Lambda_bar, Lambda_bar^LC, f=(Lbar-1)/Lambda
// ---------------------------------------------------------------------------
__global__ void k_prep_small(const float* __restrict__ Lre, const float* __restrict__ Lim,
                             const float* __restrict__ log_step,
                             float* __restrict__ lam, float* __restrict__ apow,
                             float* __restrict__ fbuf) {
  int p = blockIdx.x * blockDim.x + threadIdx.x;
  if (p >= PDIM) return;
  float lr = Lre[p], li = Lim[p];
  float st = __expf(log_step[p]);
  float er = __expf(lr * st);
  float ar = er * __cosf(li * st);
  float ai = er * __sinf(li * st);
  lam[2 * p] = ar; lam[2 * p + 1] = ai;
  float eL = __expf(lr * st * (float)LC);
  apow[2 * p]     = eL * __cosf(li * st * (float)LC);
  apow[2 * p + 1] = eL * __sinf(li * st * (float)LC);
  float nr = ar - 1.0f, ni = ai;
  float den = lr * lr + li * li;
  fbuf[2 * p]     = (nr * lr + ni * li) / den;
  fbuf[2 * p + 1] = (ni * lr - nr * li) / den;
}

// ---------------------------------------------------------------------------
// k_prep_weights: build N-major bf16 weights.
// ---------------------------------------------------------------------------
__global__ void k_prep_weights(const float* __restrict__ Bre, const float* __restrict__ Bim,
                               const float* __restrict__ Cre, const float* __restrict__ Cim,
                               const float* __restrict__ fbuf,
                               unsigned short* __restrict__ W1t,
                               unsigned short* __restrict__ W2t) {
  int idx = blockIdx.x * blockDim.x + threadIdx.x;      // over P*H
  if (idx >= PDIM * HDIM) return;
  int p = idx / HDIM, h = idx % HDIM;
  float fr = fbuf[2 * p], fi = fbuf[2 * p + 1];
  float br = Bre[(size_t)p * HDIM + h], bi = Bim[(size_t)p * HDIM + h];
  W1t[(size_t)p * HDIM + h]          = f2bf(fr * br - fi * bi);
  W1t[(size_t)(PDIM + p) * HDIM + h] = f2bf(fr * bi + fi * br);
  float cr = Cre[(size_t)h * PDIM + p], ci = Cim[(size_t)h * PDIM + p];
  W2t[(size_t)h * K2 + p]        = f2bf( 2.0f * cr);
  W2t[(size_t)h * K2 + PDIM + p] = f2bf(-2.0f * ci);
}

// ---------------------------------------------------------------------------
// k_transpose_u: (B,H,L) f32 -> (B,L,H) bf16 via 32x32 LDS tiles
// ---------------------------------------------------------------------------
__global__ __launch_bounds__(256) void k_transpose_u(const float* __restrict__ in,
                                                     unsigned short* __restrict__ ubf) {
  __shared__ float tile[32][33];
  int t  = blockIdx.x;
  int lt = t & (LDIM / 32 - 1);  t >>= 8;   // 256 l-tiles
  int ht = t & (HDIM / 32 - 1);             // 16 h-tiles
  int b  = t >> 4;
  int l0 = lt * 32, h0 = ht * 32;
  int tx = threadIdx.x & 31, ty = threadIdx.x >> 5;    // ty 0..7
  const float* src = in + ((size_t)b * HDIM + h0) * LDIM + l0;
#pragma unroll
  for (int r = 0; r < 4; ++r)
    tile[ty + r * 8][tx] = src[(size_t)(ty + r * 8) * LDIM + tx];
  __syncthreads();
  unsigned short* dst = ubf + ((size_t)b * LDIM + l0) * HDIM + h0;
#pragma unroll
  for (int r = 0; r < 4; ++r) {
    int row = ty + r * 8;                               // l offset
    dst[(size_t)row * HDIM + tx] = f2bf(tile[tx][row]);
  }
}

// ---------------------------------------------------------------------------
// WMMA macro-tile 128(M) x 64(N), K staged 32/iter via TDM double buffering.
// LDS element layout: A0 [0,5120)  A1 [5120,10240)
//                     W0 [10240,12800)  W1 [12800,15360)   rows of 40 elems.
// wave = wm*2+wn; wm in 0..3 (M), wn in 0..1 (N); 2x2 wmma per wave.
// ---------------------------------------------------------------------------
#define LDS_ELEMS 15360

__device__ __forceinline__ void gemm_tile(const unsigned short* __restrict__ A,
                                          const unsigned short* __restrict__ Wt,
                                          int m0, int n0,
                                          unsigned short* lds, v8f acc[2][2]) {
  const int tid  = threadIdx.x;
  const int lane = tid & 31;
  const int wave = tid >> 5;
  const int wm = wave >> 1, wn = wave & 1;
  const int lm = lane & 15, hi = lane >> 4;
  const unsigned ldsBase = (unsigned)(uintptr_t)(void*)lds;  // LDS byte offset
  const u32x8 g1A = tdm_group1(128);
  const u32x8 g1W = tdm_group1(64);

  if (wave == 0) {                      // prologue: fill buffer 0
    tdm_issue(&A [(size_t)m0 * K2], ldsBase,          g1A);
    tdm_issue(&Wt[(size_t)n0 * K2], ldsBase + 20480u, g1W);
  }
  for (int it = 0; it < K2 / 32; ++it) {
    const int cur = it & 1;
    if (wave == 0) {
      if (it + 1 < K2 / 32) {           // prefetch next K panel into other buffer
        const unsigned nxt = (unsigned)(cur ^ 1);
        const int k0n = (it + 1) * 32;
        tdm_issue(&A [(size_t)m0 * K2 + k0n], ldsBase + nxt * 10240u,          g1A);
        tdm_issue(&Wt[(size_t)n0 * K2 + k0n], ldsBase + 20480u + nxt * 5120u,  g1W);
        __builtin_amdgcn_s_wait_tensorcnt(2);   // current buffer's 2 TDMs done
      } else {
        __builtin_amdgcn_s_wait_tensorcnt(0);
      }
    }
    __syncthreads();                    // publish LDS buffer `cur`
    const unsigned short* At = lds + cur * 5120;
    const unsigned short* Bt = lds + 10240 + cur * 2560;
    FragBF a[2], b[2];
#pragma unroll
    for (int i = 0; i < 2; ++i) {       // A: lane holds row M, K = hi*8.. / 16+hi*8..
      int R = (wm * 32 + i * 16 + lm) * 40;
      a[i].h[0] = *(const u16x8*)&At[R + hi * 8];
      a[i].h[1] = *(const u16x8*)&At[R + 16 + hi * 8];
    }
#pragma unroll
    for (int j = 0; j < 2; ++j) {       // B(N-major): lane holds col N, K = hi*8.. / 16+hi*8..
      int R = (wn * 32 + j * 16 + lm) * 40;
      b[j].h[0] = *(const u16x8*)&Bt[R + hi * 8];
      b[j].h[1] = *(const u16x8*)&Bt[R + 16 + hi * 8];
    }
#pragma unroll
    for (int i = 0; i < 2; ++i)
#pragma unroll
      for (int j = 0; j < 2; ++j)
        acc[i][j] = __builtin_amdgcn_wmma_f32_16x16x32_bf16(
            false, a[i].v, false, b[j].v, (short)0, acc[i][j], false, false);
    __syncthreads();                    // done reading; next iter may overwrite
  }
}

// ---------------------------------------------------------------------------
// GEMM1: Bu = u_bf @ W1t^T    (M=B*L, N=2P, f32 out)
// ---------------------------------------------------------------------------
__global__ __launch_bounds__(256) void k_gemm1(const unsigned short* __restrict__ Ubf,
                                               const unsigned short* __restrict__ W1t,
                                               float* __restrict__ Bu) {
  __shared__ unsigned short lds[LDS_ELEMS];
  int nt = blockIdx.x & 7;          // 512/64 = 8 N-tiles
  int mt = blockIdx.x >> 3;         // 512 M-tiles
  int m0 = mt * 128, n0 = nt * 64;
  v8f acc[2][2] = {};
  gemm_tile(Ubf, W1t, m0, n0, lds, acc);
  const int lane = threadIdx.x & 31, wave = threadIdx.x >> 5;
  const int wm = wave >> 1, wn = wave & 1;
  const int lm = lane & 15, hi = lane >> 4;
#pragma unroll
  for (int i = 0; i < 2; ++i)
#pragma unroll
    for (int j = 0; j < 2; ++j) {
      int mB = m0 + wm * 32 + i * 16 + hi * 8;
      int n  = n0 + wn * 32 + j * 16 + lm;
#pragma unroll
      for (int r = 0; r < 8; ++r)
        Bu[(size_t)(mB + r) * K2 + n] = acc[i][j][r];
    }
}

// ---------------------------------------------------------------------------
// Scan pass 1: per (b, segment, p) local recurrence, store segment-end value.
// ---------------------------------------------------------------------------
__global__ void k_scan1(const float* __restrict__ Bu, const float* __restrict__ lam,
                        float* __restrict__ carry) {
  int t = blockIdx.x * blockDim.x + threadIdx.x;   // B*NSEG*P = 131072
  int p = t & (PDIM - 1);
  int s = (t >> 8) & (NSEG - 1);
  int b = t >> 14;
  float ar = lam[2 * p], ai = lam[2 * p + 1];
  float xr = 0.f, xi = 0.f;
  const float* base = Bu + ((size_t)b * LDIM + s * LC) * K2 + p;
#pragma unroll 4
  for (int j = 0; j < LC; ++j) {
    float br = base[(size_t)j * K2];
    float bi = base[(size_t)j * K2 + PDIM];
    float nxr = fmaf(ar, xr, fmaf(-ai, xi, br));
    float nxi = fmaf(ar, xi, fmaf( ai, xr, bi));
    xr = nxr; xi = nxi;
  }
  size_t ci = (((size_t)b * PDIM + p) * NSEG + s) * 2;
  carry[ci] = xr; carry[ci + 1] = xi;
}

// ---------------------------------------------------------------------------
// Scan pass 2: exclusive scan of segment carries (coeff = Lbar^LC), in place.
// ---------------------------------------------------------------------------
__global__ void k_scan2(const float* __restrict__ apow, float* __restrict__ carry) {
  int t = blockIdx.x * blockDim.x + threadIdx.x;   // B*P = 2048
  int p = t & (PDIM - 1);
  int b = t >> 8;
  float Ar = apow[2 * p], Ai = apow[2 * p + 1];
  float cr = 0.f, ci = 0.f;
  float* c = carry + ((size_t)b * PDIM + p) * NSEG * 2;
  for (int s = 0; s < NSEG; ++s) {
    float er = c[2 * s], ei = c[2 * s + 1];
    c[2 * s] = cr; c[2 * s + 1] = ci;                // carry-in for segment s
    float nr = fmaf(Ar, cr, fmaf(-Ai, ci, er));
    float ni = fmaf(Ar, ci, fmaf( Ai, cr, ei));
    cr = nr; ci = ni;
  }
}

// ---------------------------------------------------------------------------
// Scan pass 3: recompute local scan with carry-in, emit xs as bf16 (B*L, 2P).
// ---------------------------------------------------------------------------
__global__ void k_scan3(const float* __restrict__ Bu, const float* __restrict__ lam,
                        const float* __restrict__ carry, unsigned short* __restrict__ xbf) {
  int t = blockIdx.x * blockDim.x + threadIdx.x;
  int p = t & (PDIM - 1);
  int s = (t >> 8) & (NSEG - 1);
  int b = t >> 14;
  float ar = lam[2 * p], ai = lam[2 * p + 1];
  size_t ci = (((size_t)b * PDIM + p) * NSEG + s) * 2;
  float xr = carry[ci], xi = carry[ci + 1];
  const float* base = Bu + ((size_t)b * LDIM + s * LC) * K2 + p;
  unsigned short* ob = xbf + ((size_t)b * LDIM + s * LC) * K2 + p;
#pragma unroll 4
  for (int j = 0; j < LC; ++j) {
    float br = base[(size_t)j * K2];
    float bi = base[(size_t)j * K2 + PDIM];
    float nxr = fmaf(ar, xr, fmaf(-ai, xi, br));
    float nxi = fmaf(ar, xi, fmaf( ai, xr, bi));
    xr = nxr; xi = nxi;
    ob[(size_t)j * K2]        = f2bf(xr);
    ob[(size_t)j * K2 + PDIM] = f2bf(xi);
  }
}

// ---------------------------------------------------------------------------
// GEMM2 + D*u + exact GELU, output directly in (B,H,L).
// Each lane's C fragment = 8 consecutive l at fixed h -> contiguous stores.
// ---------------------------------------------------------------------------
__global__ __launch_bounds__(256) void k_gemm2(const unsigned short* __restrict__ Xbf,
                                               const unsigned short* __restrict__ W2t,
                                               const float* __restrict__ u_in,
                                               const float* __restrict__ D,
                                               float* __restrict__ out) {
  __shared__ unsigned short lds[LDS_ELEMS];
  int nt = blockIdx.x & 7;
  int mt = blockIdx.x >> 3;
  int m0 = mt * 128, n0 = nt * 64;
  v8f acc[2][2] = {};
  gemm_tile(Xbf, W2t, m0, n0, lds, acc);
  const int lane = threadIdx.x & 31, wave = threadIdx.x >> 5;
  const int wm = wave >> 1, wn = wave & 1;
  const int lm = lane & 15, hi = lane >> 4;
#pragma unroll
  for (int i = 0; i < 2; ++i)
#pragma unroll
    for (int j = 0; j < 2; ++j) {
      int m = m0 + wm * 32 + i * 16 + hi * 8;          // global M = b*L + l
      int b = m >> 13;                                  // L = 8192
      int l = m & (LDIM - 1);
      int h = n0 + wn * 32 + j * 16 + lm;
      float dh = D[h];
      size_t off = ((size_t)b * HDIM + h) * LDIM + l;
      const float* up = u_in + off;
      float*       op = out  + off;
#pragma unroll
      for (int r = 0; r < 8; ++r) {
        float y = acc[i][j][r] + dh * up[r];
        y = 0.5f * y * (1.0f + erff(y * 0.70710678118654752f));  // exact GELU
        op[r] = y;
      }
    }
}

// ---------------------------------------------------------------------------
extern "C" void kernel_launch(void* const* d_in, const int* in_sizes, int n_in,
                              void* d_out, int out_size, void* d_ws, size_t ws_size,
                              hipStream_t stream) {
  (void)in_sizes; (void)n_in; (void)out_size; (void)ws_size;
  const float* u_in = (const float*)d_in[0];   // (B,1,H,1,L)
  const float* Lre  = (const float*)d_in[1];
  const float* Lim  = (const float*)d_in[2];
  const float* Bre  = (const float*)d_in[3];
  const float* Bim  = (const float*)d_in[4];
  const float* Cre  = (const float*)d_in[5];
  const float* Cim  = (const float*)d_in[6];
  const float* D    = (const float*)d_in[7];
  const float* lstp = (const float*)d_in[8];
  float* out = (float*)d_out;

  char* ws = (char*)d_ws;
  unsigned short* ubf  = (unsigned short*)(ws + OFF_UBF);
  unsigned short* W1t  = (unsigned short*)(ws + OFF_W1T);
  unsigned short* W2t  = (unsigned short*)(ws + OFF_W2T);
  float*          lam  = (float*)(ws + OFF_LAM);
  float*          apow = (float*)(ws + OFF_APOW);
  float*          fbuf = (float*)(ws + OFF_F);
  float*          carr = (float*)(ws + OFF_CARRY);
  float*          Bu   = (float*)(ws + OFF_BU);
  unsigned short* xbf  = (unsigned short*)(ws + OFF_XSBF);

  k_prep_small<<<1, 256, 0, stream>>>(Lre, Lim, lstp, lam, apow, fbuf);
  k_prep_weights<<<(PDIM * HDIM) / 256, 256, 0, stream>>>(Bre, Bim, Cre, Cim, fbuf, W1t, W2t);
  k_transpose_u<<<B_SZ * (HDIM / 32) * (LDIM / 32), 256, 0, stream>>>(u_in, ubf);
  k_gemm1<<<(B_SZ * LDIM / 128) * (K2 / 64), 256, 0, stream>>>(ubf, W1t, Bu);
  k_scan1<<<(B_SZ * NSEG * PDIM) / 256, 256, 0, stream>>>(Bu, lam, carr);
  k_scan2<<<(B_SZ * PDIM) / 256, 256, 0, stream>>>(apow, carr);
  k_scan3<<<(B_SZ * NSEG * PDIM) / 256, 256, 0, stream>>>(Bu, lam, carr, xbf);
  k_gemm2<<<(B_SZ * LDIM / 128) * (HDIM / 64), 256, 0, stream>>>(xbf, W2t, u_in, D, out);
}
